// Causal_Temporal_Map_Attention_2_66700842107384
// MI455X (gfx1250) — matrix operand order repaired and checked by reading, not compiled, not run on hardware
//
#include <hip/hip_runtime.h>

// ---------------------------------------------------------------------------
// out = SCALE * (x_e Wq^T) @ ((x_e Wk^T)^T @ x)      [no softmax -> reassociate]
// All GEMMs in bf16 via v_wmma_f32_16x16x32_bf16 (wave32, CDNA5).
// Tiles staged into double-buffered LDS with GLOBAL_LOAD_ASYNC_TO_LDS_B128
// (ASYNCcnt) where the layout is contiguous; register+scatter otherwise.
// ---------------------------------------------------------------------------

typedef __bf16 bf16;
typedef __attribute__((ext_vector_type(16))) __bf16 bf16x16;
typedef __attribute__((ext_vector_type(8)))  __bf16 bf16x8;
typedef __attribute__((ext_vector_type(8)))  float  floatx8;
typedef __attribute__((ext_vector_type(4)))  int    i32x4;

#define DIM_X 256
#define DIM_E 256
#define DHID  512                 // DIM_X + DIM_E
#define BATCH 8
#define SEQ   2048
#define MROWS (BATCH * SEQ)       // 16384
#define LDSP  40                  // 32 + 8 pad -> 80B rows, 16B-aligned b128 access

__device__ __forceinline__ bf16 to_bf16(float f) {
  union { float f; unsigned u; } in; in.f = f;
  unsigned u = in.u;
  u += 0x7FFFu + ((u >> 16) & 1u);          // round-to-nearest-even
  union { unsigned short s; bf16 h; } out;
  out.s = (unsigned short)(u >> 16);
  return out.h;
}

// Async 16-byte global -> LDS copy (per-lane), tracked by ASYNCcnt.
__device__ __forceinline__ void async_ld16(bf16* lds, const bf16* g) {
#if __has_builtin(__builtin_amdgcn_global_load_async_to_lds_b128)
  typedef __attribute__((address_space(1))) i32x4 gi32x4;   // global v4i
  typedef __attribute__((address_space(3))) i32x4 li32x4;   // LDS v4i
  __builtin_amdgcn_global_load_async_to_lds_b128((gi32x4*)g, (li32x4*)lds, 0, 0);
#else
  const unsigned           loff = (unsigned)(unsigned long long)lds;  // low 32 = LDS offset
  const unsigned long long ga   = (unsigned long long)g;
  asm volatile("global_load_async_to_lds_b128 %0, %1, off"
               :: "v"(loff), "v"(ga) : "memory");
#endif
}

__device__ __forceinline__ void wait_async0() {
#if __has_builtin(__builtin_amdgcn_s_wait_asynccnt)
  __builtin_amdgcn_s_wait_asynccnt(0);
#else
  asm volatile("s_wait_asynccnt 0x0" ::: "memory");
#endif
}

// A fragment: 16x32 bf16. Lane l (half=l>>4, lr=l&15) holds row M=row0+lr,
// K = half*8 + {0..7} in v[0..7] and 16 + half*8 + {0..7} in v[8..15].
__device__ __forceinline__ bf16x16 load_a_frag(const bf16* As, int row0, int lr, int half) {
  const bf16* p = As + (row0 + lr) * LDSP + half * 8;
  union { bf16x16 v; bf16x8 h[2]; } u;
  u.h[0] = *reinterpret_cast<const bf16x8*>(p);
  u.h[1] = *reinterpret_cast<const bf16x8*>(p + 16);
  return u.v;
}

// B fragment: 32x16 bf16. Lane l holds N=col0+(l&15), K = half*8 + {0..7} in v[0..7] and 16 + half*8 + {0..7} in v[8..15].
// Bs is stored N-major: Bs[n*LDSP + k].
__device__ __forceinline__ bf16x16 load_b_frag(const bf16* Bs, int col0, int lr, int half) {
  const bf16* p = Bs + (col0 + lr) * LDSP + half * 8;
  union { bf16x16 v; bf16x8 h[2]; } u;
  u.h[0] = *reinterpret_cast<const bf16x8*>(p);
  u.h[1] = *reinterpret_cast<const bf16x8*>(p + 16);
  return u.v;
}

// C = scale * A * B.
//   A_KMAJOR=false: A[m,k] at A[m*lda + k]   (row-major)       -> async path
//   A_KMAJOR=true : A[m,k] at A[k*lda + m]   (K-major, K^T x)  -> reg+scatter
//   B_KMAJOR=false: B[k,n] at B[n*ldb + k]   (N-major weights) -> async path
//   B_KMAJOR=true : B[k,n] at B[k*ldb + n]   (row-major K x N) -> reg+scatter
// Block: 128x128 tile, BK=32, 256 thr = 8 waves in 4x2 grid, 32x64 per wave.
// Double-buffered LDS, software pipelined.
template <bool A_KMAJOR, bool B_KMAJOR, bool OUT_BF16>
__global__ __launch_bounds__(256)
void gemm_wmma_bf16(const bf16* __restrict__ A, const bf16* __restrict__ B,
                    void* __restrict__ Cv, int K, int lda, int ldb, int ldc,
                    long asb, long bsb, long csb, float scale)
{
  __shared__ bf16 As[2][128 * LDSP];
  __shared__ bf16 Bs[2][128 * LDSP];

  const int  tid  = threadIdx.x;
  const long bz   = blockIdx.z;
  A += bz * asb;
  B += bz * bsb;
  const int rowT = blockIdx.y * 128;
  const int colT = blockIdx.x * 128;
  const int wid  = tid >> 5;
  const int lane = tid & 31;
  const int wr   = wid >> 1;      // 0..3 (32 rows each)
  const int wc   = wid & 1;       // 0..1 (64 cols each)
  const int half = lane >> 4;
  const int lr   = lane & 15;

  // ---- per-thread staging addresses (hoisted; advance by fixed stride) ----
  const bf16* ga[2]; int la[2];
  const bf16* gb[2]; int lb[2];
#pragma unroll
  for (int p = 0; p < 2; ++p) {
    const int linear = p * 256 + tid;
    if (!A_KMAJOR) {
      const int m = linear >> 2, kc = (linear & 3) << 3;
      ga[p] = A + (long)(rowT + m) * lda + kc;
      la[p] = m * LDSP + kc;
    } else {
      const int k = linear >> 4, ms = (linear & 15) << 3;
      ga[p] = A + (long)k * lda + rowT + ms;
      la[p] = ms * LDSP + k;
    }
    if (!B_KMAJOR) {
      const int n = linear >> 2, kc = (linear & 3) << 3;
      gb[p] = B + (long)(colT + n) * ldb + kc;
      lb[p] = n * LDSP + kc;
    } else {
      const int k = linear >> 4, ns = (linear & 15) << 3;
      gb[p] = B + (long)k * ldb + colT + ns;
      lb[p] = ns * LDSP + k;
    }
  }
  const long stepA = A_KMAJOR ? (long)32 * lda : 32;
  const long stepB = B_KMAJOR ? (long)32 * ldb : 32;

  bf16x8 sva[2], svb[2];   // register staging for the scatter (transposed) paths

  auto stage_async = [&](int buf) {    // issue async global->LDS, contiguous ops
#pragma unroll
    for (int p = 0; p < 2; ++p) {
      if (!A_KMAJOR) async_ld16(&As[buf][la[p]], ga[p]);
      if (!B_KMAJOR) async_ld16(&Bs[buf][lb[p]], gb[p]);
    }
  };
  auto stage_load = [&]() {            // issue global->reg loads, transposed ops
#pragma unroll
    for (int p = 0; p < 2; ++p) {
      if (A_KMAJOR) sva[p] = *reinterpret_cast<const bf16x8*>(ga[p]);
      if (B_KMAJOR) svb[p] = *reinterpret_cast<const bf16x8*>(gb[p]);
    }
  };
  auto stage_commit = [&](int buf) {   // transpose-scatter regs into LDS
#pragma unroll
    for (int p = 0; p < 2; ++p) {
      if (A_KMAJOR) {
#pragma unroll
        for (int j = 0; j < 8; ++j) As[buf][la[p] + j * LDSP] = sva[p][j];
      }
      if (B_KMAJOR) {
#pragma unroll
        for (int j = 0; j < 8; ++j) Bs[buf][lb[p] + j * LDSP] = svb[p][j];
      }
    }
  };
  auto stage_advance = [&]() {
#pragma unroll
    for (int p = 0; p < 2; ++p) { ga[p] += stepA; gb[p] += stepB; }
  };

  const floatx8 zero = {0.f, 0.f, 0.f, 0.f, 0.f, 0.f, 0.f, 0.f};
  floatx8 acc[2][4];
#pragma unroll
  for (int mi = 0; mi < 2; ++mi)
#pragma unroll
    for (int ni = 0; ni < 4; ++ni) acc[mi][ni] = zero;

  // ---- prologue: fill buffer 0 ----
  stage_async(0);
  stage_load();
  stage_commit(0);
  stage_advance();
  if (!A_KMAJOR || !B_KMAJOR) wait_async0();
  __syncthreads();

  const int kIters = K >> 5;
  int cb = 0;
  for (int it = 0; it < kIters; ++it) {
    const bool more = (it + 1 < kIters);
    // stage tile i+1 into the other buffer while computing tile i
    if (more) { stage_async(cb ^ 1); stage_load(); }

    bf16x16 af[2], bfg[4];
#pragma unroll
    for (int mi = 0; mi < 2; ++mi) af[mi]  = load_a_frag(&As[cb][0], wr * 32 + mi * 16, lr, half);
#pragma unroll
    for (int ni = 0; ni < 4; ++ni) bfg[ni] = load_b_frag(&Bs[cb][0], wc * 64 + ni * 16, lr, half);

#pragma unroll
    for (int mi = 0; mi < 2; ++mi)
#pragma unroll
      for (int ni = 0; ni < 4; ++ni)
        acc[mi][ni] = __builtin_amdgcn_wmma_f32_16x16x32_bf16(
            false, af[mi], false, bfg[ni], (short)0, acc[mi][ni], false, false);

    if (more) { stage_commit(cb ^ 1); stage_advance(); }
    if ((!A_KMAJOR || !B_KMAJOR) && more) wait_async0();
    __syncthreads();
    cb ^= 1;
  }

  // ---- epilogue: D VGPR r -> row = r + half*8, col = lr within 16x16 tile ----
#pragma unroll
  for (int mi = 0; mi < 2; ++mi)
#pragma unroll
    for (int ni = 0; ni < 4; ++ni)
#pragma unroll
      for (int r = 0; r < 8; ++r) {
        const int  row = rowT + wr * 32 + mi * 16 + half * 8 + r;
        const int  col = colT + wc * 64 + ni * 16 + lr;
        const float v = acc[mi][ni][r] * scale;
        if (OUT_BF16) {
          bf16* C = reinterpret_cast<bf16*>(Cv) + bz * csb;
          C[(long)row * ldc + col] = to_bf16(v);
        } else {
          float* C = reinterpret_cast<float*>(Cv) + bz * csb;
          C[(long)row * ldc + col] = v;
        }
      }
}

// Build x_e = concat(x, broadcast(e)) in bf16; also emit x in bf16 (the "V").
__global__ void build_xe_kernel(const float* __restrict__ x, const float* __restrict__ e,
                                bf16* __restrict__ xe, bf16* __restrict__ xb)
{
  const long idx = (long)blockIdx.x * blockDim.x + threadIdx.x;  // < MROWS*DHID
  const int  t   = (int)(idx & (DHID - 1));
  const long m   = idx >> 9;
  const int  n   = (int)(m & (SEQ - 1));
  float v;
  if (t < DIM_X) v = x[m * DIM_X + t];
  else           v = e[(long)n * DIM_E + (t - DIM_X)];
  const bf16 h = to_bf16(v);
  xe[idx] = h;
  if (t < DIM_X) xb[m * DIM_X + t] = h;
}

__global__ void cvt_bf16_kernel(const float* __restrict__ in, bf16* __restrict__ out, int n)
{
  const int i = blockIdx.x * blockDim.x + threadIdx.x;
  if (i < n) out[i] = to_bf16(in[i]);
}

extern "C" void kernel_launch(void* const* d_in, const int* in_sizes, int n_in,
                              void* d_out, int out_size, void* d_ws, size_t ws_size,
                              hipStream_t stream)
{
  (void)in_sizes; (void)n_in; (void)out_size; (void)ws_size;
  const float* x  = (const float*)d_in[0];   // (8, 2048, 256)
  const float* e  = (const float*)d_in[1];   // (2048, 256)
  const float* Wq = (const float*)d_in[2];   // (512, 512) torch (out,in)
  const float* Wk = (const float*)d_in[3];   // (512, 512)

  bf16* ws  = (bf16*)d_ws;
  bf16* xe  = ws;                                 // 16384*512
  bf16* xb  = xe  + (size_t)MROWS * DHID;         // 16384*256
  bf16* wqb = xb  + (size_t)MROWS * DIM_X;        // 512*512 (torch layout = N-major B)
  bf16* wkb = wqb + (size_t)DHID * DHID;          // 512*512
  bf16* Qb  = wkb + (size_t)DHID * DHID;          // 16384*512
  bf16* Kb  = Qb  + (size_t)MROWS * DHID;         // 16384*512
  bf16* M3  = Kb  + (size_t)MROWS * DHID;         // 8*512*256

  build_xe_kernel<<<(MROWS * DHID) / 256, 256, 0, stream>>>(x, e, xe, xb);
  cvt_bf16_kernel<<<(DHID * DHID) / 256, 256, 0, stream>>>(Wq, wqb, DHID * DHID);
  cvt_bf16_kernel<<<(DHID * DHID) / 256, 256, 0, stream>>>(Wk, wkb, DHID * DHID);

  // Q = x_e @ Wq^T   (B = Wq in (out,in) layout == N-major) : fully async-staged
  gemm_wmma_bf16<false, false, true>
      <<<dim3(DHID / 128, MROWS / 128, 1), 256, 0, stream>>>(
          xe, wqb, Qb, DHID, DHID, DHID, DHID, 0L, 0L, 0L, 1.0f);
  // K = x_e @ Wk^T
  gemm_wmma_bf16<false, false, true>
      <<<dim3(DHID / 128, MROWS / 128, 1), 256, 0, stream>>>(
          xe, wkb, Kb, DHID, DHID, DHID, DHID, 0L, 0L, 0L, 1.0f);
  // M3_b = K_b^T @ x_b : (512 x 256) per batch.  A[m=dk,k=n] = K_b[n*512+dk]
  gemm_wmma_bf16<true, true, true>
      <<<dim3(DIM_X / 128, DHID / 128, BATCH), 256, 0, stream>>>(
          Kb, xb, M3, SEQ, DHID, DIM_X, DIM_X,
          (long)SEQ * DHID, (long)SEQ * DIM_X, (long)DHID * DIM_X, 1.0f);
  // out_b = SCALE * Q_b @ M3_b : (2048 x 256) per batch, f32 output.
  gemm_wmma_bf16<false, true, false>
      <<<dim3(DIM_X / 128, SEQ / 128, BATCH), 256, 0, stream>>>(
          Qb, M3, d_out, DHID, DHID, DIM_X, DIM_X,
          (long)SEQ * DHID, (long)DHID * DIM_X, (long)SEQ * DIM_X,
          0.044194173824159216f /* (DIM_X+DIM_E)^-0.5 */);
}
